// TemporalGraphTransformer_24086176596040
// MI455X (gfx1250) — matrix operand order repaired and checked by reading, not compiled, not run on hardware
//
#include <hip/hip_runtime.h>
#include <hip/hip_bf16.h>
#include <math.h>

typedef __attribute__((ext_vector_type(16))) _Float16 v16h;
typedef __attribute__((ext_vector_type(8)))  float    v8f;

#define TT   100
#define HIDN 128
#define NHEAD 4

static __device__ __forceinline__ int lane_id() { return threadIdx.x & 31; }
static __device__ __forceinline__ float sigmoidf(float v) { return 1.f / (1.f + __expf(-v)); }

// ---- WMMA fragment packing per CDNA5 ISA 7.12.2 (wave32) ----
// A (16Mx32K f16): lane = M (mod 16); element h -> K = pairpattern + 8*(lane/16).
static __device__ __forceinline__ v16h packA_f32(const float* src, int ld, int m0, int k0, int mlim) {
  int lane = lane_id();
  int m = m0 + (lane & 15);
  int g = lane >> 4;
  v16h r;
  if (m >= mlim) {
#pragma unroll
    for (int h = 0; h < 16; ++h) r[h] = (_Float16)0.f;
    return r;
  }
  const float* p = src + (size_t)m * ld + k0;
#pragma unroll
  for (int h = 0; h < 16; ++h) {
    int v = h >> 1;
    int kk = (v < 4 ? 2 * v : 16 + 2 * (v - 4)) + 8 * g + (h & 1);
    r[h] = (_Float16)p[kk];
  }
  return r;
}

// A fragment from native-f16 row-major source (two contiguous 16B chunks per lane).
static __device__ __forceinline__ v16h packA16(const _Float16* src, int ld, int m0, int k0, int mlim) {
  int lane = lane_id();
  int m = m0 + (lane & 15);
  int g = lane >> 4;
  v16h r;
  if (m >= mlim) {
#pragma unroll
    for (int h = 0; h < 16; ++h) r[h] = (_Float16)0.f;
    return r;
  }
  const _Float16* p = src + (size_t)m * ld + k0;
#pragma unroll
  for (int h = 0; h < 16; ++h) {
    int v = h >> 1;
    int kk = (v < 4 ? 2 * v : 16 + 2 * (v - 4)) + 8 * g + (h & 1);
    r[h] = p[kk];
  }
  return r;
}

// A fragment with node->[t][b] remap (rows stored [T][B][256], f16)
static __device__ __forceinline__ v16h packA16_node(const _Float16* l1, int Bb, int m0, int k0) {
  int lane = lane_id();
  int m = m0 + (lane & 15);
  int g = lane >> 4;
  int t = m % TT, b = m / TT;
  const _Float16* p = l1 + ((size_t)t * Bb + b) * 256 + k0;
  v16h r;
#pragma unroll
  for (int h = 0; h < 16; ++h) {
    int v = h >> 1;
    int kk = (v < 4 ? 2 * v : 16 + 2 * (v - 4)) + 8 * g + (h & 1);
    r[h] = p[kk];
  }
  return r;
}

// B fragment from pre-converted f16 weights: B[k][n] = W[n][k], W row-major [O][ld].
// Lane = N; K = (h&7) + 16*(h>>3) + 8*(lane/16): two 16B chunks per lane.
static __device__ __forceinline__ v16h packB16(const _Float16* W, int ld, int k0, int o0) {
  int lane = lane_id();
  int o = o0 + (lane & 15);
  int g = lane >> 4;
  const _Float16* p = W + (size_t)o * ld + k0 + 8 * g;
  v16h r;
#pragma unroll
  for (int h = 0; h < 16; ++h) r[h] = p[(h & 7) + 16 * (h >> 3)];
  return r;
}

static __device__ __forceinline__ v8f wmma_f16(v16h a, v16h b, v8f c) {
  return __builtin_amdgcn_wmma_f32_16x16x32_f16(false, a, false, b, (short)0, c, false, false);
}

// ---------------- Kernel 0a: one-time f32 -> f16 weight conversion ----------------
__global__ void k_f32tof16(const float* __restrict__ s, _Float16* __restrict__ d, int n) {
  int i = blockIdx.x * 256 + threadIdx.x;
  if (i < n) d[i] = (_Float16)s[i];
}

// ---------------- Kernel 0b: mh16 = agg + gat_b (f16 materialization for final GEMM) ----------------
__global__ void k_addbias16(const float* __restrict__ agg, const float* __restrict__ gatb,
                            _Float16* __restrict__ mh, int n) {
  int i = blockIdx.x * 256 + threadIdx.x;
  if (i < n) mh[i] = (_Float16)(agg[i] + gatb[i & (HIDN - 1)]);
}

// ---------------- Kernel 1: input projection + positional encoding ----------------
// One wave: 16 node-rows x full 128 cols (A fragment reused across 8 WMMAs).
__global__ void k_inproj(const float* __restrict__ x, const _Float16* __restrict__ Win,
                         const float* __restrict__ bin, _Float16* __restrict__ h0,
                         int N, int Bb) {
  int wid = blockIdx.x * 8 + (threadIdx.x >> 5);
  if (wid >= N / 16) return;
  int m0 = wid * 16;
  v8f acc[8] = {};
#pragma unroll
  for (int ks = 0; ks < 2; ++ks) {
    v16h a = packA_f32(x, 64, m0, ks * 32, N);
#pragma unroll
    for (int q = 0; q < 8; ++q) {
      v16h b = packB16(Win, 64, ks * 32, q * 16);
      acc[q] = wmma_f16(a, b, acc[q]);
    }
  }
  int lane = lane_id();
  int g = lane >> 4;
  int n15 = lane & 15;
  float bb[8], freq[8];
#pragma unroll
  for (int q = 0; q < 8; ++q) {
    int col = q * 16 + n15;
    bb[q] = bin[col];
    freq[q] = __expf(-(float)(col & ~1) * (9.210340372f / 128.f));
  }
#pragma unroll
  for (int r = 0; r < 8; ++r) {
    int row = m0 + r + 8 * g;
    int t = row % TT, b = row / TT;
    size_t base = ((size_t)t * Bb + b) * HIDN;
#pragma unroll
    for (int q = 0; q < 8; ++q) {
      int col = q * 16 + n15;
      float pe = (col & 1) ? __cosf((float)t * freq[q]) : __sinf((float)t * freq[q]);
      h0[base + col] = (_Float16)(acc[q][r] + bb[q] + pe);
    }
  }
}

// ---------------- Kernel 2: one fused BiLSTM timestep ----------------
// f16 weights + f16 h-state feed WMMA directly (no cvt on the serial path);
// gate nonlinearity fused; ping-pong state; c-state kept in f32.
template <int IND>
__global__ void k_lstm_step(const _Float16* __restrict__ xin, const _Float16* __restrict__ Wih,
                            const _Float16* __restrict__ Whh, const float* __restrict__ bih,
                            const float* __restrict__ bhh, _Float16* __restrict__ hst,
                            float* __restrict__ cst, _Float16* __restrict__ lout,
                            int t, int Bb, int pin) {
  int dir = blockIdx.z;
  int teff = dir ? (TT - 1 - t) : t;
  const _Float16* Wih_d = Wih + (size_t)dir * 512 * IND;
  const _Float16* Whh_d = Whh + (size_t)dir * 512 * 128;
  const float* bih_d = bih + dir * 512;
  const float* bhh_d = bhh + dir * 512;
  const _Float16* h_in = hst + (size_t)(pin * 2 + dir) * Bb * 128;
  const float* c_in = cst + (size_t)(pin * 2 + dir) * Bb * 128;
  _Float16* h_out = hst + (size_t)((pin ^ 1) * 2 + dir) * Bb * 128;
  float* c_out = cst + (size_t)((pin ^ 1) * 2 + dir) * Bb * 128;
  const _Float16* xin_t = xin + (size_t)teff * Bb * IND;

  int m0 = blockIdx.x * 16;
  int cidx = threadIdx.x >> 5;
  v8f acc[4] = {};
#pragma unroll
  for (int ks = 0; ks < IND / 32; ++ks) {
    v16h a = packA16(xin_t, IND, m0, ks * 32, Bb);
#pragma unroll
    for (int q = 0; q < 4; ++q) {
      v16h b = packB16(Wih_d, IND, ks * 32, q * 128 + cidx * 16);
      acc[q] = wmma_f16(a, b, acc[q]);
    }
  }
#pragma unroll
  for (int ks = 0; ks < 4; ++ks) {
    v16h a = packA16(h_in, 128, m0, ks * 32, Bb);
#pragma unroll
    for (int q = 0; q < 4; ++q) {
      v16h b = packB16(Whh_d, 128, ks * 32, q * 128 + cidx * 16);
      acc[q] = wmma_f16(a, b, acc[q]);
    }
  }
  int lane = lane_id();
  int g = lane >> 4;
  int j = cidx * 16 + (lane & 15);
  float bi0 = bih_d[j] + bhh_d[j];
  float bf0 = bih_d[128 + j] + bhh_d[128 + j];
  float bg0 = bih_d[256 + j] + bhh_d[256 + j];
  float bo0 = bih_d[384 + j] + bhh_d[384 + j];
#pragma unroll
  for (int r = 0; r < 8; ++r) {
    int b = m0 + r + 8 * g;
    if (b < Bb) {
      float iv = acc[0][r] + bi0;
      float fv = acc[1][r] + bf0;
      float gv = acc[2][r] + bg0;
      float ov = acc[3][r] + bo0;
      float cn = sigmoidf(fv) * c_in[(size_t)b * 128 + j] + sigmoidf(iv) * tanhf(gv);
      float hn = sigmoidf(ov) * tanhf(cn);
      c_out[(size_t)b * 128 + j] = cn;
      _Float16 hn16 = (_Float16)hn;
      h_out[(size_t)b * 128 + j] = hn16;
      lout[((size_t)teff * Bb + b) * 256 + dir * 128 + j] = hn16;
    }
  }
}

// ---------------- Kernel 3: GAT z = h1 @ W_all^T (4 heads stacked) ----------------
// One wave: 16 node-rows x full 128 cols, A (f16 l1) loaded once per k-step.
__global__ void k_zproj(const _Float16* __restrict__ l1, const _Float16* __restrict__ gatW,
                        float* __restrict__ z, int N, int Bb) {
  int wid = blockIdx.x * 8 + (threadIdx.x >> 5);
  if (wid >= N / 16) return;
  int m0 = wid * 16;
  v8f acc[8] = {};
#pragma unroll
  for (int ks = 0; ks < 8; ++ks) {
    v16h a = packA16_node(l1, Bb, m0, ks * 32);
#pragma unroll
    for (int q = 0; q < 8; ++q) {
      v16h b = packB16(gatW, 256, ks * 32, q * 16);
      acc[q] = wmma_f16(a, b, acc[q]);
    }
  }
  int lane = lane_id();
  int g = lane >> 4;
  int n15 = lane & 15;
#pragma unroll
  for (int r = 0; r < 8; ++r) {
    int row = m0 + r + 8 * g;
#pragma unroll
    for (int q = 0; q < 8; ++q)
      z[(size_t)row * 128 + q * 16 + n15] = acc[q][r];
  }
}

// ---------------- Kernel 4: per-node attention coefficients ----------------
__global__ void k_attcoef(const float* __restrict__ z, const float* __restrict__ asrc,
                          const float* __restrict__ adst, float* __restrict__ es,
                          float* __restrict__ ed, int N) {
  int tid = blockIdx.x * blockDim.x + threadIdx.x;
  if (tid >= N * NHEAD) return;
  int n = tid >> 2, h = tid & 3;
  const float* zp = z + (size_t)n * 128 + h * 32;
  const float* as = asrc + h * 32;
  const float* ad = adst + h * 32;
  float s1 = 0.f, s2 = 0.f;
#pragma unroll
  for (int k = 0; k < 32; ++k) { float v = zp[k]; s1 += v * as[k]; s2 += v * ad[k]; }
  es[tid] = s1;
  ed[tid] = s2;
}

static __device__ __forceinline__ float lrelu2(float v) { return v > 0.f ? v : 0.2f * v; }
static __device__ __forceinline__ unsigned fkey(float f) {
  unsigned u = __float_as_uint(f);
  return (u & 0x80000000u) ? ~u : (u | 0x80000000u);
}
static __device__ __forceinline__ float fdec(unsigned key) {
  return __uint_as_float((key & 0x80000000u) ? (key ^ 0x80000000u) : ~key);
}

// ---------------- Kernel 5: segment max over dst ----------------
__global__ void k_edgemax(const int* __restrict__ ei, const float* __restrict__ es,
                          const float* __restrict__ ed, unsigned* __restrict__ maxk,
                          int E, int EP) {
  int tid = blockIdx.x * blockDim.x + threadIdx.x;
  if (tid >= EP * NHEAD) return;
  int h = tid & 3, e = tid >> 2;
  int s = (e < E) ? ei[e] : (e - E);
  int d = (e < E) ? ei[E + e] : (e - E);
  float lv = lrelu2(es[s * NHEAD + h] + ed[d * NHEAD + h]);
  atomicMax(&maxk[d * NHEAD + h], fkey(lv));
}

// ---------------- Kernel 6: softmax denominator ----------------
__global__ void k_edgeden(const int* __restrict__ ei, const float* __restrict__ es,
                          const float* __restrict__ ed, const unsigned* __restrict__ maxk,
                          float* __restrict__ den, int E, int EP) {
  int tid = blockIdx.x * blockDim.x + threadIdx.x;
  if (tid >= EP * NHEAD) return;
  int h = tid & 3, e = tid >> 2;
  int s = (e < E) ? ei[e] : (e - E);
  int d = (e < E) ? ei[E + e] : (e - E);
  float lv = lrelu2(es[s * NHEAD + h] + ed[d * NHEAD + h]);
  float m = fdec(maxk[d * NHEAD + h]);
  atomicAdd(&den[d * NHEAD + h], __expf(lv - m));
}

// ---------------- Kernel 7: weighted aggregation (wave per (edge,head), lane per channel) ----------------
__global__ void k_edgeagg(const int* __restrict__ ei, const float* __restrict__ es,
                          const float* __restrict__ ed, const unsigned* __restrict__ maxk,
                          const float* __restrict__ den, const float* __restrict__ z,
                          float* __restrict__ agg, int E, int EP) {
  int tid = blockIdx.x * blockDim.x + threadIdx.x;
  int k = tid & 31;
  int pair = tid >> 5;
  if (pair >= EP * NHEAD) return;
  int h = pair & 3, e = pair >> 2;
  int s = (e < E) ? ei[e] : (e - E);
  int d = (e < E) ? ei[E + e] : (e - E);
  float lv = lrelu2(es[s * NHEAD + h] + ed[d * NHEAD + h]);
  float m = fdec(maxk[d * NHEAD + h]);
  float alpha = __expf(lv - m) / den[d * NHEAD + h];
  atomicAdd(&agg[(size_t)d * 128 + h * 32 + k], alpha * z[(size_t)s * 128 + h * 32 + k]);
}

// ---------------- Kernel 8: final projection + fused in-register LayerNorm ----------------
__global__ void k_final(const _Float16* __restrict__ mh16, const _Float16* __restrict__ projW,
                        const float* __restrict__ projb, const float* __restrict__ lng,
                        const float* __restrict__ lnb, float* __restrict__ out, int N) {
  int wid = blockIdx.x * 8 + (threadIdx.x >> 5);
  if (wid >= N / 16) return;
  int m0 = wid * 16;
  v8f acc[8] = {};
#pragma unroll
  for (int ks = 0; ks < 4; ++ks) {
    v16h a = packA16(mh16, 128, m0, ks * 32, N);
#pragma unroll
    for (int q = 0; q < 8; ++q) {
      v16h b = packB16(projW, 128, ks * 32, q * 16);
      acc[q] = wmma_f16(a, b, acc[q]);
    }
  }
  int lane = lane_id();
  int g = lane >> 4;
  int n15 = lane & 15;
#pragma unroll
  for (int r = 0; r < 8; ++r) {
    float vals[8];
    float s = 0.f, sq = 0.f;
#pragma unroll
    for (int q = 0; q < 8; ++q) {
      float v = acc[q][r] + projb[q * 16 + n15];
      vals[q] = v; s += v; sq += v * v;
    }
#pragma unroll
    for (int off = 1; off < 16; off <<= 1) {
      s += __shfl_xor(s, off, 32);
      sq += __shfl_xor(sq, off, 32);
    }
    float mean = s * (1.f / 128.f);
    float var = sq * (1.f / 128.f) - mean * mean;
    float inv = rsqrtf(var + 1e-5f);
    int row = m0 + r + 8 * g;
#pragma unroll
    for (int q = 0; q < 8; ++q) {
      int col = q * 16 + n15;
      out[(size_t)row * 128 + col] = (vals[q] - mean) * inv * lng[col] + lnb[col];
    }
  }
}

extern "C" void kernel_launch(void* const* d_in, const int* in_sizes, int n_in,
                              void* d_out, int out_size, void* d_ws, size_t ws_size,
                              hipStream_t stream) {
  (void)n_in; (void)out_size; (void)ws_size;
  const float* x    = (const float*)d_in[0];
  const int*   ei   = (const int*)d_in[1];
  const float* Win  = (const float*)d_in[3];
  const float* bin  = (const float*)d_in[4];
  const float* Wih0 = (const float*)d_in[5];
  const float* Whh0 = (const float*)d_in[6];
  const float* bih0 = (const float*)d_in[7];
  const float* bhh0 = (const float*)d_in[8];
  const float* Wih1 = (const float*)d_in[9];
  const float* Whh1 = (const float*)d_in[10];
  const float* bih1 = (const float*)d_in[11];
  const float* bhh1 = (const float*)d_in[12];
  const float* gatW = (const float*)d_in[13];
  const float* asrc = (const float*)d_in[14];
  const float* adst = (const float*)d_in[15];
  const float* gatb = (const float*)d_in[16];
  const float* projW = (const float*)d_in[17];
  const float* projb = (const float*)d_in[18];
  const float* lng  = (const float*)d_in[19];
  const float* lnb  = (const float*)d_in[20];
  float* out = (float*)d_out;

  int N  = in_sizes[0] / 64;   // 50000
  int E  = in_sizes[1] / 2;    // 800000
  int Bb = N / TT;             // 500
  int EP = E + N;

  // ---- workspace layout (256B-aligned chunks) ----
  char* wp = (char*)d_ws;
  auto alloc = [&](size_t bytes) -> char* {
    char* r = wp;
    wp += (bytes + 255) & ~(size_t)255;
    return r;
  };
  _Float16* h0  = (_Float16*)alloc((size_t)N * 128 * 2);      // [T][B][128] f16
  _Float16* l0  = (_Float16*)alloc((size_t)N * 256 * 2);      // [T][B][256] f16
  _Float16* l1  = (_Float16*)alloc((size_t)N * 256 * 2);      // [T][B][256] f16
  float*    zb  = (float*)alloc((size_t)N * 128 * 4);         // [N][128]
  float*    agg = (float*)alloc((size_t)N * 128 * 4);         // [N][128]
  _Float16* mh16 = (_Float16*)alloc((size_t)N * 128 * 2);     // agg + gat_b, f16
  float*    esb = (float*)alloc((size_t)N * 4 * 4);
  float*    edb = (float*)alloc((size_t)N * 4 * 4);
  float*    den = (float*)alloc((size_t)N * 4 * 4);
  unsigned* maxk = (unsigned*)alloc((size_t)N * 4 * 4);
  _Float16* hst = (_Float16*)alloc((size_t)2 * 2 * Bb * 128 * 2);  // ping-pong f16 h
  float*    cst = (float*)alloc((size_t)2 * 2 * Bb * 128 * 4);     // ping-pong f32 c
  _Float16* winh  = (_Float16*)alloc((size_t)128 * 64 * 2);
  _Float16* wih0h = (_Float16*)alloc((size_t)2 * 512 * 128 * 2);
  _Float16* whh0h = (_Float16*)alloc((size_t)2 * 512 * 128 * 2);
  _Float16* wih1h = (_Float16*)alloc((size_t)2 * 512 * 256 * 2);
  _Float16* whh1h = (_Float16*)alloc((size_t)2 * 512 * 128 * 2);
  _Float16* gatwh = (_Float16*)alloc((size_t)128 * 256 * 2);
  _Float16* projwh = (_Float16*)alloc((size_t)128 * 128 * 2);

  // ---- one-time weight f32->f16 conversion (L2-resident thereafter) ----
  k_f32tof16<<<dim3((128 * 64 + 255) / 256), 256, 0, stream>>>(Win, winh, 128 * 64);
  k_f32tof16<<<dim3((2 * 512 * 128 + 255) / 256), 256, 0, stream>>>(Wih0, wih0h, 2 * 512 * 128);
  k_f32tof16<<<dim3((2 * 512 * 128 + 255) / 256), 256, 0, stream>>>(Whh0, whh0h, 2 * 512 * 128);
  k_f32tof16<<<dim3((2 * 512 * 256 + 255) / 256), 256, 0, stream>>>(Wih1, wih1h, 2 * 512 * 256);
  k_f32tof16<<<dim3((2 * 512 * 128 + 255) / 256), 256, 0, stream>>>(Whh1, whh1h, 2 * 512 * 128);
  k_f32tof16<<<dim3((128 * 256 + 255) / 256), 256, 0, stream>>>(gatW, gatwh, 128 * 256);
  k_f32tof16<<<dim3((128 * 128 + 255) / 256), 256, 0, stream>>>(projW, projwh, 128 * 128);

  hipMemsetAsync(maxk, 0, (size_t)N * 4 * sizeof(unsigned), stream);
  hipMemsetAsync(den,  0, (size_t)N * 4 * sizeof(float), stream);
  hipMemsetAsync(agg,  0, (size_t)N * 128 * sizeof(float), stream);

  // input projection + positional encoding (f16 activation out)
  k_inproj<<<dim3((N / 16 + 7) / 8), 256, 0, stream>>>(x, winh, bin, h0, N, Bb);

  size_t hbytes = (size_t)2 * 2 * Bb * 128 * sizeof(_Float16);
  size_t cbytes = (size_t)2 * 2 * Bb * 128 * sizeof(float);
  int mtiles = (Bb + 15) / 16;

  // BiLSTM layer 0
  hipMemsetAsync(hst, 0, hbytes, stream);
  hipMemsetAsync(cst, 0, cbytes, stream);
  for (int t = 0; t < TT; ++t)
    k_lstm_step<128><<<dim3(mtiles, 1, 2), 256, 0, stream>>>(h0, wih0h, whh0h, bih0, bhh0,
                                                             hst, cst, l0, t, Bb, t & 1);
  // BiLSTM layer 1
  hipMemsetAsync(hst, 0, hbytes, stream);
  hipMemsetAsync(cst, 0, cbytes, stream);
  for (int t = 0; t < TT; ++t)
    k_lstm_step<256><<<dim3(mtiles, 1, 2), 256, 0, stream>>>(l0, wih1h, whh1h, bih1, bhh1,
                                                             hst, cst, l1, t, Bb, t & 1);

  // GAT
  k_zproj<<<dim3((N / 16 + 7) / 8), 256, 0, stream>>>(l1, gatwh, zb, N, Bb);
  k_attcoef<<<dim3((N * 4 + 255) / 256), 256, 0, stream>>>(zb, asrc, adst, esb, edb, N);
  k_edgemax<<<dim3((EP * 4 + 255) / 256), 256, 0, stream>>>(ei, esb, edb, maxk, E, EP);
  k_edgeden<<<dim3((EP * 4 + 255) / 256), 256, 0, stream>>>(ei, esb, edb, maxk, den, E, EP);
  k_edgeagg<<<dim3((EP * 4 * 32 + 255) / 256), 256, 0, stream>>>(ei, esb, edb, maxk, den, zb, agg, E, EP);

  // projection + LayerNorm
  k_addbias16<<<dim3((N * 128 + 255) / 256), 256, 0, stream>>>(agg, gatb, mh16, N * 128);
  k_final<<<dim3((N / 16 + 7) / 8), 256, 0, stream>>>(mh16, projwh, projb, lng, lnb, out, N);
}
